// MultiHeadAttentionWithRelativePosition_71176198029850
// MI455X (gfx1250) — matrix operand order repaired and checked by reading, not compiled, not run on hardware
//
#include <hip/hip_runtime.h>
#include <hip/hip_bf16.h>

typedef _Float16 v16h __attribute__((ext_vector_type(16)));
typedef _Float16 v8h  __attribute__((ext_vector_type(8)));
typedef float    v8f  __attribute__((ext_vector_type(8)));

#define DM     1024
#define NH     16
#define DKH    64
#define SLEN   2048
#define BATCH  2
#define MAXREL 32
#define NEGINF (-1.0e9f)

static __device__ __forceinline__ v16h cat8(v8h lo, v8h hi) {
  return __builtin_shufflevector(lo, hi, 0,1,2,3,4,5,6,7,8,9,10,11,12,13,14,15);
}

// A-fragment (16xK=32, f16). A is row-major with leading dim ld.
static __device__ __forceinline__ v16h load_a(const _Float16* A, size_t ld,
                                              int row0, int k0, int lane) {
  int m    = lane & 15;
  int koff = (lane >> 4) * 8;
  const _Float16* p = A + (size_t)(row0 + m) * ld + (size_t)(k0 + koff);
  v8h lo = *(const v8h*)p;
  v8h hi = *(const v8h*)(p + 16);
  return cat8(lo, hi);
}

// B-fragment (K=32 x 16, f16) where B[k][n] = W[col0+n][k], W row-major, ld = leading dim.
static __device__ __forceinline__ v16h load_b(const _Float16* W, size_t ld,
                                              int col0, int k0, int lane) {
  int n  = lane & 15;
  int kb = (lane >> 4) * 8;
  const _Float16* p = W + (size_t)(col0 + n) * ld + (size_t)(k0 + kb);
  v8h lo = *(const v8h*)p;
  v8h hi = *(const v8h*)(p + 16);
  return cat8(lo, hi);
}

static __device__ __forceinline__ v8f wmma32(v16h a, v16h b, v8f c) {
  return __builtin_amdgcn_wmma_f32_16x16x32_f16(false, a, false, b, (short)0, c,
                                                false, false);
}

// ---------------------------------------------------------------------------
// fp32 -> f16 conversion
// ---------------------------------------------------------------------------
__global__ void cvt_f32_to_f16(const float* __restrict__ src,
                               _Float16* __restrict__ dst, int n) {
  int i = blockIdx.x * blockDim.x + threadIdx.x;
  int stride = gridDim.x * blockDim.x;
  for (; i < n; i += stride) dst[i] = (_Float16)src[i];
}

// ---------------------------------------------------------------------------
// WMMA GEMM: Y = A @ W^T (+bias)*scale.  Software-pipelined k-loop.
// Each wave: 32x64 register tile (2 A-frags x 4 B-frags = 8 WMMAs per k-step).
// Block: 256 threads = 8 waves, arranged 2(M) x 4(N): block tile 64x256.
// mode 0: f16 out [b][h][s][dk]; mode 1: f16 out [b][h][dk][s]; mode 2: f32 row-major.
// ---------------------------------------------------------------------------
__global__ __launch_bounds__(256)
void gemm_wmma_kernel(const _Float16* __restrict__ A, const _Float16* __restrict__ W,
                      const float* __restrict__ bias, float scale, int mode,
                      _Float16* __restrict__ out16, float* __restrict__ out32) {
  int lane = threadIdx.x & 31;
  int wave = threadIdx.x >> 5;
  int row0 = blockIdx.x * 64  + (wave & 1) * 32;
  int col0 = blockIdx.y * 256 + (wave >> 1) * 64;

  v8f acc0[4] = {};
  v8f acc1[4] = {};

  // prologue: fragments for k = 0
  v16h a0 = load_a(A, DM, row0,      0, lane);
  v16h a1 = load_a(A, DM, row0 + 16, 0, lane);
  v16h b0 = load_b(W, DM, col0,      0, lane);
  v16h b1 = load_b(W, DM, col0 + 16, 0, lane);
  v16h b2 = load_b(W, DM, col0 + 32, 0, lane);
  v16h b3 = load_b(W, DM, col0 + 48, 0, lane);

  for (int k = 32; k < DM; k += 32) {
    // preload next k-step while current WMMAs execute
    v16h na0 = load_a(A, DM, row0,      k, lane);
    v16h na1 = load_a(A, DM, row0 + 16, k, lane);
    v16h nb0 = load_b(W, DM, col0,      k, lane);
    v16h nb1 = load_b(W, DM, col0 + 16, k, lane);
    v16h nb2 = load_b(W, DM, col0 + 32, k, lane);
    v16h nb3 = load_b(W, DM, col0 + 48, k, lane);

    acc0[0] = wmma32(a0, b0, acc0[0]);
    acc0[1] = wmma32(a0, b1, acc0[1]);
    acc0[2] = wmma32(a0, b2, acc0[2]);
    acc0[3] = wmma32(a0, b3, acc0[3]);
    acc1[0] = wmma32(a1, b0, acc1[0]);
    acc1[1] = wmma32(a1, b1, acc1[1]);
    acc1[2] = wmma32(a1, b2, acc1[2]);
    acc1[3] = wmma32(a1, b3, acc1[3]);

    a0 = na0; a1 = na1;
    b0 = nb0; b1 = nb1; b2 = nb2; b3 = nb3;
  }
  // epilogue k-step
  acc0[0] = wmma32(a0, b0, acc0[0]);
  acc0[1] = wmma32(a0, b1, acc0[1]);
  acc0[2] = wmma32(a0, b2, acc0[2]);
  acc0[3] = wmma32(a0, b3, acc0[3]);
  acc1[0] = wmma32(a1, b0, acc1[0]);
  acc1[1] = wmma32(a1, b1, acc1[1]);
  acc1[2] = wmma32(a1, b2, acc1[2]);
  acc1[3] = wmma32(a1, b3, acc1[3]);

  int rowoff = (lane >> 4) * 8;
  int lcol   = lane & 15;
#pragma unroll
  for (int ms = 0; ms < 2; ++ms) {
    const v8f* acc = ms ? acc1 : acc0;
    int rbase = row0 + ms * 16 + rowoff;
#pragma unroll
    for (int t = 0; t < 4; ++t) {
      int col = col0 + t * 16 + lcol;
      float bc = bias[col];
#pragma unroll
      for (int r = 0; r < 8; ++r) {
        int row = rbase + r;
        float v = (acc[t][r] + bc) * scale;
        if (mode == 0) {
          int bb = row >> 11, s = row & (SLEN - 1);
          int h = col >> 6,  dk = col & (DKH - 1);
          out16[(((size_t)bb * NH + h) * SLEN + s) * DKH + dk] = (_Float16)v;
        } else if (mode == 1) {
          int bb = row >> 11, s = row & (SLEN - 1);
          int h = col >> 6,  dk = col & (DKH - 1);
          out16[(((size_t)bb * NH + h) * DKH + dk) * SLEN + s] = (_Float16)v;
        } else {
          out32[(size_t)row * DM + col] = v;
        }
      }
    }
  }
}

// ---------------------------------------------------------------------------
// Flash attention with relative-position bias + causal + padding mask.
// Grid: (S/64, NH, B). Block: 128 threads = 4 waves; each wave owns 16 queries.
// ---------------------------------------------------------------------------
struct alignas(16) WaveLds {
  _Float16 p[16][40];   // P tile (16q x 32k) f16, row stride 40 halfs (16B aligned)
  float    sc[16][33];  // score tile f32, padded
  float    alpha[16];
  float    linv[16];
};

__global__ __launch_bounds__(128)
void attn_kernel(const _Float16* __restrict__ Q, const _Float16* __restrict__ K,
                 const _Float16* __restrict__ Vt, const float* __restrict__ rel_emb,
                 const int* __restrict__ mask, _Float16* __restrict__ ctx) {
  __shared__ WaveLds wl[4];
  __shared__ float rel_h[2 * MAXREL + 1];

  int h = blockIdx.y, b = blockIdx.z;
  for (int i = threadIdx.x; i < 2 * MAXREL + 1; i += 128)
    rel_h[i] = rel_emb[(size_t)i * NH + h];
  __syncthreads();

  int lane   = threadIdx.x & 31;
  int wave   = threadIdx.x >> 5;
  int qb     = blockIdx.x * 64;
  int qw     = qb + wave * 16;
  int lcol   = lane & 15;
  int rowoff = (lane >> 4) * 8;

  const _Float16* Qh = Q  + ((size_t)b * NH + h) * SLEN * DKH;
  const _Float16* Kh = K  + ((size_t)b * NH + h) * SLEN * DKH;
  const _Float16* Vh = Vt + ((size_t)b * NH + h) * DKH * SLEN;
  const int*      mb = mask + (size_t)b * SLEN;

  v16h a0 = load_a(Qh, DKH, qw, 0,  lane);
  v16h a1 = load_a(Qh, DKH, qw, 32, lane);

  v8f acc[4] = {};
  float mrun = -3.0e38f, lrun = 0.f;

  int kend = qb + 64;  // causal bound, uniform across the block
  for (int kb = 0; kb < kend; kb += 32) {
    // prefetch next key/value tiles into cache across the softmax bubble
    if (kb + 32 < kend) {
      __builtin_prefetch(Kh + (size_t)(kb + 32 + lane) * DKH, 0, 0);
      __builtin_prefetch(Vh + (size_t)lane * SLEN + (kb + 32), 0, 0);
      __builtin_prefetch(Vh + (size_t)(lane + 32) * SLEN + (kb + 32), 0, 0);
    }

    // ---- issue all K and V fragment loads up front ----
    v16h kb00 = load_b(Kh, DKH, kb,      0,  lane);
    v16h kb01 = load_b(Kh, DKH, kb,      32, lane);
    v16h kb10 = load_b(Kh, DKH, kb + 16, 0,  lane);
    v16h kb11 = load_b(Kh, DKH, kb + 16, 32, lane);
    v16h vb0  = load_b(Vh, SLEN, 0,  kb, lane);
    v16h vb1  = load_b(Vh, SLEN, 16, kb, lane);
    v16h vb2  = load_b(Vh, SLEN, 32, kb, lane);
    v16h vb3  = load_b(Vh, SLEN, 48, kb, lane);

    // ---- scores: 16q x 32k, contraction over d=64 ----
    v8f s0 = {};
    s0 = wmma32(a0, kb00, s0);
    s0 = wmma32(a1, kb01, s0);
    v8f s1 = {};
    s1 = wmma32(a0, kb10, s1);
    s1 = wmma32(a1, kb11, s1);

    int k0g = kb + lcol, k1g = kb + 16 + lcol;
    bool pad0 = mb[k0g] != 0, pad1 = mb[k1g] != 0;
#pragma unroll
    for (int r = 0; r < 8; ++r) {
      int q = qw + rowoff + r;
      int d0 = min(max(k0g - q, -MAXREL), MAXREL);
      float v0 = s0[r] + rel_h[d0 + MAXREL];
      if (k0g > q || !pad0) v0 = NEGINF;
      wl[wave].sc[rowoff + r][lcol] = v0;
      int d1 = min(max(k1g - q, -MAXREL), MAXREL);
      float v1 = s1[r] + rel_h[d1 + MAXREL];
      if (k1g > q || !pad1) v1 = NEGINF;
      wl[wave].sc[rowoff + r][16 + lcol] = v1;
    }
    __syncthreads();

    // ---- online softmax: lanes 0..15 each own one query row ----
    if (lane < 16) {
      const float* row = wl[wave].sc[lane];
      float tmax = row[0];
#pragma unroll
      for (int j = 1; j < 32; ++j) tmax = fmaxf(tmax, row[j]);
      float mnew = fmaxf(mrun, tmax);
      float al   = __expf(mrun - mnew);
      float sum  = 0.f;
#pragma unroll
      for (int j = 0; j < 32; ++j) {
        float pv = __expf(row[j] - mnew);
        sum += pv;
        wl[wave].p[lane][j] = (_Float16)pv;
      }
      lrun = lrun * al + sum;
      mrun = mnew;
      wl[wave].alpha[lane] = al;
    }
    __syncthreads();

    // ---- rescale accumulators by alpha[row] ----
    const float* ap = wl[wave].alpha + rowoff;
#pragma unroll
    for (int r = 0; r < 8; ++r) {
      float a = ap[r];
      acc[0][r] *= a; acc[1][r] *= a; acc[2][r] *= a; acc[3][r] *= a;
    }

    // ---- P @ V: A-fragment from LDS, B-fragments already in registers ----
    int koff = (lane >> 4) * 8;
    const _Float16* pp = &wl[wave].p[lcol][0];
    v16h pa = cat8(*(const v8h*)(pp + koff), *(const v8h*)(pp + 16 + koff));
    acc[0] = wmma32(pa, vb0, acc[0]);
    acc[1] = wmma32(pa, vb1, acc[1]);
    acc[2] = wmma32(pa, vb2, acc[2]);
    acc[3] = wmma32(pa, vb3, acc[3]);
  }

  // ---- epilogue: divide by l, store ctx as [b][s][h*64+dk] f16 ----
  if (lane < 16) wl[wave].linv[lane] = 1.0f / lrun;
  __syncthreads();
  const float* lp = wl[wave].linv + rowoff;
#pragma unroll
  for (int r = 0; r < 8; ++r) {
    float inv = lp[r];
    int q = qw + rowoff + r;
    _Float16* op = ctx + ((size_t)b * SLEN + q) * DM + h * DKH + lcol;
#pragma unroll
    for (int t = 0; t < 4; ++t) op[t * 16] = (_Float16)(acc[t][r] * inv);
  }
}

// ---------------------------------------------------------------------------
// Host launcher
// ---------------------------------------------------------------------------
extern "C" void kernel_launch(void* const* d_in, const int* in_sizes, int n_in,
                              void* d_out, int out_size, void* d_ws, size_t ws_size,
                              hipStream_t stream) {
  (void)in_sizes; (void)n_in; (void)out_size; (void)ws_size;
  const float* x      = (const float*)d_in[0];
  const int*   mask   = (const int*)  d_in[1];
  const float* Wq     = (const float*)d_in[2];
  const float* bq     = (const float*)d_in[3];
  const float* Wk     = (const float*)d_in[4];
  const float* bk     = (const float*)d_in[5];
  const float* Wv     = (const float*)d_in[6];
  const float* bv     = (const float*)d_in[7];
  const float* Wo     = (const float*)d_in[8];
  const float* bo     = (const float*)d_in[9];
  const float* relemb = (const float*)d_in[10];
  float* out = (float*)d_out;

  const size_t XSZ = (size_t)BATCH * SLEN * DM;  // 4,194,304
  const size_t WSZ = (size_t)DM * DM;            // 1,048,576

  _Float16* w     = (_Float16*)d_ws;
  _Float16* x16   = w; w += XSZ;
  _Float16* wq16  = w; w += WSZ;
  _Float16* wk16  = w; w += WSZ;
  _Float16* wv16  = w; w += WSZ;
  _Float16* wo16  = w; w += WSZ;
  _Float16* q16   = w; w += XSZ;
  _Float16* k16   = w; w += XSZ;
  _Float16* vt16  = w; w += XSZ;
  _Float16* ctx16 = w; w += XSZ;

  cvt_f32_to_f16<<<256, 256, 0, stream>>>(x,  x16,  (int)XSZ);
  cvt_f32_to_f16<<<64,  256, 0, stream>>>(Wq, wq16, (int)WSZ);
  cvt_f32_to_f16<<<64,  256, 0, stream>>>(Wk, wk16, (int)WSZ);
  cvt_f32_to_f16<<<64,  256, 0, stream>>>(Wv, wv16, (int)WSZ);
  cvt_f32_to_f16<<<64,  256, 0, stream>>>(Wo, wo16, (int)WSZ);

  dim3 gg(BATCH * SLEN / 64, DM / 256);  // (64, 4)
  gemm_wmma_kernel<<<gg, 256, 0, stream>>>(x16, wq16, bq, 0.125f, 0, q16,  nullptr);
  gemm_wmma_kernel<<<gg, 256, 0, stream>>>(x16, wk16, bk, 1.0f,   0, k16,  nullptr);
  gemm_wmma_kernel<<<gg, 256, 0, stream>>>(x16, wv16, bv, 1.0f,   1, vt16, nullptr);

  dim3 ga(SLEN / 64, NH, BATCH);  // (32, 16, 2)
  attn_kernel<<<ga, 128, 0, stream>>>(q16, k16, vt16, relemb, mask, ctx16);

  gemm_wmma_kernel<<<gg, 256, 0, stream>>>(ctx16, wo16, bo, 1.0f, 2, nullptr, out);
}
